// RecurrentTransformerEncoder_489626272056
// MI455X (gfx1250) — compile-verified, matrix operand order checked by reading, not run on hardware
//
#include <hip/hip_runtime.h>

// ---- model dims ----
#define NB   64      // batch
#define NT   16      // segments
#define NSEG 50      // segment length
#define NE   512     // embed dim
#define ND   512     // encoder dim
#define NH   8       // heads
#define NDH  64      // head dim
#define NF   2048    // ffn dim
#define NM   (NB*NSEG)   // rows per segment = 3200
#define ATTN_SCALE 0.125f  // 1/sqrt(64)

typedef _Float16 v16h __attribute__((ext_vector_type(16)));
typedef _Float16 v8h  __attribute__((ext_vector_type(8)));
typedef _Float16 v4h  __attribute__((ext_vector_type(4)));
typedef float    v8f  __attribute__((ext_vector_type(8)));
typedef float    v4f  __attribute__((ext_vector_type(4)));

// ---------------------------------------------------------------------------
// Weight prep: W[K,N] f32  ->  Wt[N,K] f16 (WMMA B operand loads contiguous
// 16B chunks per lane, mirroring the 16-bit A-matrix lane layout).
// ---------------------------------------------------------------------------
__global__ __launch_bounds__(256) void convert_transpose_kernel(
    const float* __restrict__ W, _Float16* __restrict__ Wt, int K, int N)
{
    size_t i = (size_t)blockIdx.x * 256 + threadIdx.x;
    size_t total = (size_t)K * N;
    if (i >= total) return;
    int k = (int)(i / N);
    int n = (int)(i % N);
    Wt[(size_t)n * K + k] = (_Float16)W[i];
}

// ---------------------------------------------------------------------------
// Activation prep: f32 -> f16, elementwise (vector-4). Done ONCE per GEMM
// input instead of once per n-tile inside the GEMM (32-128x less cvt VALU).
// ---------------------------------------------------------------------------
__global__ __launch_bounds__(256) void cvt_f16_kernel(
    const float* __restrict__ A, _Float16* __restrict__ O, int n4)
{
    int i = blockIdx.x * 256 + threadIdx.x;
    if (i >= n4) return;
    v4f x = *(const v4f*)(A + (size_t)i * 4);
    v4h y;
#pragma unroll
    for (int j = 0; j < 4; ++j) y[j] = (_Float16)x[j];
    *(v4h*)(O + (size_t)i * 4) = y;
}

// ---------------------------------------------------------------------------
// Gather segment t of seqs [NB, NT*NSEG, NE] into X [NM, NE] (contiguous rows)
// ---------------------------------------------------------------------------
__global__ __launch_bounds__(256) void gather_seg_kernel(
    const float* __restrict__ seqs, float* __restrict__ X, int t)
{
    int row = blockIdx.x;                 // 0..NM-1
    int b = row / NSEG, s = row % NSEG;
    const float* src = seqs + ((size_t)b * (NT * NSEG) + (size_t)t * NSEG + s) * NE;
    float* dst = X + (size_t)row * NE;
    dst[threadIdx.x]       = src[threadIdx.x];
    dst[threadIdx.x + 256] = src[threadIdx.x + 256];
}

// ---------------------------------------------------------------------------
// WMMA GEMM: C[M,N] f32 = A[M,K] f16 @ Bt ([N,K] f16).  Optional fused ReLU.
//
// Block = 256 threads = 8 wave32, covering a 128(M) x 64(N) super-tile.
// Each wave owns one 16-row m-tile and FOUR 16x16 accumulators (4 n-tiles),
// reusing its A fragment 4x.  The 64x32-half B chunk per k-step is staged
// once per block into LDS with gfx1250 async copy
// (global_load_async_to_lds_b128, double-buffered, s_wait_asynccnt) and
// shared by all 8 waves via ds_load_b128 -> 8x less B global traffic.
//
// Software pipelining inside the k-step: all 8 ds_load_b128 B-fragment loads
// are issued BEFORE the 4-deep WMMA chain, so the compiler can use partial
// s_wait_dscnt values and overlap LDS latency with the matrix pipe instead
// of stalling each WMMA on its own just-issued loads.
//
// Lane layout (CDNA5 ISA 7.12.2, 16-bit A 16x32):
//   lanes 0-15 : row M=lane,   halves 0..7 -> K=k0..k0+7,  8..15 -> k0+16..23
//   lanes16-31 : row M=lane-16,halves 0..7 -> K=k0+8..15,  8..15 -> k0+24..31
// B operand symmetric with N = lane%16.  C/D: c[r] = (M=r+(lane>=16?8:0),
// N=lane%16).
// ---------------------------------------------------------------------------
__global__ __launch_bounds__(256) void gemm_wmma_f16_kernel(
    const _Float16* __restrict__ A, const _Float16* __restrict__ Bt,
    float* __restrict__ C, int M, int K, int N, int relu)
{
    __shared__ alignas(16) _Float16 bsm[2][64 * 32];   // 2 x 4KB double buffer

    int tid  = threadIdx.x;
    int wave = tid >> 5;
    int lane = tid & 31;
    int lr   = lane & 15;
    int hs   = lane >> 4;                  // K sub-offset select (0/8)
    int m0   = (blockIdx.y * 8 + wave) * 16;
    int n0   = blockIdx.x * 64;

    const _Float16* arow = A + (size_t)(m0 + lr) * K + hs * 8;
    // staging source: thread tid copies 8 halves (16B) of row n0 + tid/4,
    // k-halves (tid&3)*8 .. +7  (256 threads cover 64 rows x 32 halves)
    const _Float16* bsrc = Bt + (size_t)(n0 + (tid >> 2)) * K + (tid & 3) * 8;

    auto stage = [&](int buf, int k0) {
        unsigned lds = (unsigned)(uintptr_t)&bsm[buf][tid * 8];
        const _Float16* g = bsrc + k0;
        asm volatile("global_load_async_to_lds_b128 %0, %1, off"
                     :: "v"(lds), "v"(g) : "memory");
    };

    v8f acc[4] = {};

    stage(0, 0);
    int pb = 0;
    for (int k0 = 0; k0 < K; k0 += 32, pb ^= 1) {
        asm volatile("s_wait_asynccnt 0x0" ::: "memory");
        __syncthreads();                       // staged B chunk visible
        if (k0 + 32 < K) stage(pb ^ 1, k0 + 32);  // prefetch next chunk

        // A fragment (2 x global b128, pre-converted f16)
        v8h a0 = *(const v8h*)(arow + k0);
        v8h a1 = *(const v8h*)(arow + k0 + 16);
        v16h a;
#pragma unroll
        for (int i = 0; i < 8; ++i) { a[i] = a0[i]; a[8 + i] = a1[i]; }

        // Issue ALL B-fragment LDS loads first (8 x ds_load_b128) ...
        v8h bfrag[8];
#pragma unroll
        for (int j = 0; j < 4; ++j) {
            const _Float16* bp = &bsm[pb][(j * 16 + lr) * 32 + hs * 8];
            bfrag[2 * j]     = *(const v8h*)(bp);        // ds_load_b128
            bfrag[2 * j + 1] = *(const v8h*)(bp + 16);   // ds_load_b128
        }
        // ... then run the 4-deep WMMA chain (partial dscnt waits overlap)
#pragma unroll
        for (int j = 0; j < 4; ++j) {
            v16h b;
#pragma unroll
            for (int i = 0; i < 8; ++i) {
                b[i]     = bfrag[2 * j][i];
                b[8 + i] = bfrag[2 * j + 1][i];
            }
            acc[j] = __builtin_amdgcn_wmma_f32_16x16x32_f16(
                         false, a, false, b, (short)0, acc[j], false, false);
        }
        __syncthreads();                       // readers done before re-stage
    }

#pragma unroll
    for (int j = 0; j < 4; ++j) {
        int n = n0 + j * 16 + lr;
#pragma unroll
        for (int r = 0; r < 8; ++r) {
            int m = m0 + r + hs * 8;
            float v = acc[j][r];
            if (relu) v = v > 0.f ? v : 0.f;
            C[(size_t)m * N + n] = v;
        }
    }
}

// ---------------------------------------------------------------------------
// Attention: one block per (batch, head); one thread per query row.
// t >= 0 : encoder self-attn, key length from ends (min(diff, NSEG))
// t <  0 : connection cross-attn, key length = NSEG
// (~1.5% of total FLOPs -> plain VALU is the right choice.)
// ---------------------------------------------------------------------------
__global__ __launch_bounds__(64) void attn_kernel(
    const float* __restrict__ Q, const float* __restrict__ Km,
    const float* __restrict__ Vm, float* __restrict__ O,
    const int* __restrict__ ends, int t)
{
    int bh = blockIdx.x;
    int b = bh / NH;
    int h = bh % NH;
    int qi = threadIdx.x;
    if (qi >= NSEG) return;

    int len = NSEG;
    if (t >= 0) {
        int e1 = ends[b * NT + t];
        int e0 = (t > 0) ? ends[b * NT + t - 1] : 0;
        len = e1 - e0;
        if (len > NSEG) len = NSEG;
        if (len < 0) len = 0;
    }

    const float* qrow = Q + (size_t)(b * NSEG + qi) * ND + h * NDH;
    float qv[NDH];
#pragma unroll
    for (int d = 0; d < NDH; ++d) qv[d] = qrow[d];

    float s[NSEG];
    float mx = -1e30f;
    for (int j = 0; j < NSEG; ++j) {
        const float* krow = Km + (size_t)(b * NSEG + j) * ND + h * NDH;
        float acc = 0.f;
#pragma unroll
        for (int d = 0; d < NDH; ++d) acc += qv[d] * krow[d];
        acc *= ATTN_SCALE;
        if (j >= len) acc = -1e9f;
        s[j] = acc;
        if (acc > mx) mx = acc;
    }
    float denom = 0.f;
    for (int j = 0; j < NSEG; ++j) { float e = __expf(s[j] - mx); s[j] = e; denom += e; }
    float inv = 1.f / denom;

    float o[NDH];
#pragma unroll
    for (int d = 0; d < NDH; ++d) o[d] = 0.f;
    for (int j = 0; j < NSEG; ++j) {
        const float* vrow = Vm + (size_t)(b * NSEG + j) * ND + h * NDH;
        float p = s[j] * inv;
#pragma unroll
        for (int d = 0; d < NDH; ++d) o[d] += p * vrow[d];
    }
    float* orow = O + (size_t)(b * NSEG + qi) * ND + h * NDH;
#pragma unroll
    for (int d = 0; d < NDH; ++d) orow[d] = o[d];
}

// ---------------------------------------------------------------------------
// O[row,:] = LayerNorm(A[row,:] + R[row,:]) * G + Bb  (eps 1e-5, ddof=0)
// ---------------------------------------------------------------------------
__global__ __launch_bounds__(256) void add_ln_kernel(
    const float* __restrict__ A, const float* __restrict__ R,
    const float* __restrict__ G, const float* __restrict__ Bb,
    float* __restrict__ O)
{
    __shared__ float red[256];
    int row = blockIdx.x;
    const float* a = A + (size_t)row * ND;
    const float* r = R + (size_t)row * ND;
    int tx = threadIdx.x;
    float x0 = a[tx] + r[tx];
    float x1 = a[tx + 256] + r[tx + 256];

    red[tx] = x0 + x1;
    __syncthreads();
    for (int s = 128; s > 0; s >>= 1) {
        if (tx < s) red[tx] += red[tx + s];
        __syncthreads();
    }
    float mean = red[0] * (1.0f / ND);
    __syncthreads();
    float d0 = x0 - mean, d1 = x1 - mean;
    red[tx] = d0 * d0 + d1 * d1;
    __syncthreads();
    for (int s = 128; s > 0; s >>= 1) {
        if (tx < s) red[tx] += red[tx + s];
        __syncthreads();
    }
    float var = red[0] * (1.0f / ND);
    float rs = rsqrtf(var + 1e-5f);
    float* o = O + (size_t)row * ND;
    o[tx]       = d0 * rs * G[tx]       + Bb[tx];
    o[tx + 256] = d1 * rs * G[tx + 256] + Bb[tx + 256];
}

// ---------------------------------------------------------------------------
extern "C" void kernel_launch(void* const* d_in, const int* in_sizes, int n_in,
                              void* d_out, int out_size, void* d_ws, size_t ws_size,
                              hipStream_t stream) {
    const float* seqs = (const float*)d_in[0];
    const int*   ends = (const int*)d_in[1];
    const float* eWin = (const float*)d_in[2];
    const float* eWq  = (const float*)d_in[3];
    const float* eWk  = (const float*)d_in[4];
    const float* eWv  = (const float*)d_in[5];
    const float* eWo  = (const float*)d_in[6];
    const float* eG1  = (const float*)d_in[7];
    const float* eB1  = (const float*)d_in[8];
    const float* eF1  = (const float*)d_in[9];
    const float* eF2  = (const float*)d_in[10];
    const float* eG2  = (const float*)d_in[11];
    const float* eB2  = (const float*)d_in[12];
    const float* cWq  = (const float*)d_in[13];
    const float* cWk  = (const float*)d_in[14];
    const float* cWv  = (const float*)d_in[15];
    const float* cWo  = (const float*)d_in[16];
    const float* cG1  = (const float*)d_in[17];
    const float* cB1  = (const float*)d_in[18];
    const float* cF1  = (const float*)d_in[19];
    const float* cF2  = (const float*)d_in[20];
    const float* cG2  = (const float*)d_in[21];
    const float* cB2  = (const float*)d_in[22];
    float* out = (float*)d_out;       // [NB, NSEG, ND] — also the carried state

    // ---- workspace layout ----
    char* ws = (char*)d_ws;
    size_t off = 0;
    auto alloc = [&](size_t bytes) -> char* {
        char* p = ws + off;
        off = (off + bytes + 255) & ~(size_t)255;
        return p;
    };
    const size_t WSQ = (size_t)ND * ND * sizeof(_Float16);   // 512x512 f16
    const size_t WFF = (size_t)ND * NF * sizeof(_Float16);   // 512x2048 f16
    _Float16* wt_eWin = (_Float16*)alloc(WSQ);
    _Float16* wt_eWq  = (_Float16*)alloc(WSQ);
    _Float16* wt_eWk  = (_Float16*)alloc(WSQ);
    _Float16* wt_eWv  = (_Float16*)alloc(WSQ);
    _Float16* wt_eWo  = (_Float16*)alloc(WSQ);
    _Float16* wt_eF1  = (_Float16*)alloc(WFF);
    _Float16* wt_eF2  = (_Float16*)alloc(WFF);
    _Float16* wt_cWq  = (_Float16*)alloc(WSQ);
    _Float16* wt_cWk  = (_Float16*)alloc(WSQ);
    _Float16* wt_cWv  = (_Float16*)alloc(WSQ);
    _Float16* wt_cWo  = (_Float16*)alloc(WSQ);
    _Float16* wt_cF1  = (_Float16*)alloc(WFF);
    _Float16* wt_cF2  = (_Float16*)alloc(WFF);

    const size_t ACT = (size_t)NM * ND * sizeof(float);      // 3200x512 f32
    float* Xb    = (float*)alloc(ACT);
    float* Hb    = (float*)alloc(ACT);
    float* Qb    = (float*)alloc(ACT);
    float* Kb    = (float*)alloc(ACT);
    float* Vb    = (float*)alloc(ACT);
    float* ATTNb = (float*)alloc(ACT);
    float* T1b   = (float*)alloc(ACT);
    float* T2b   = (float*)alloc(ACT);
    float* ENCb  = (float*)alloc(ACT);
    float* MIDb  = (float*)alloc((size_t)NM * NF * sizeof(float));  // 3200x2048
    // f16 activation staging: H16A reused per-GEMM, H16B persists for Q/K/V
    _Float16* H16A = (_Float16*)alloc((size_t)NM * NF * sizeof(_Float16));
    _Float16* H16B = (_Float16*)alloc((size_t)NM * ND * sizeof(_Float16));
    if (off > ws_size) return;   // workspace too small — nothing we can do

    // ---- helpers ----
    auto conv = [&](const float* W, _Float16* Wt, int K, int N) {
        size_t total = (size_t)K * N;
        convert_transpose_kernel<<<(unsigned)((total + 255) / 256), 256, 0, stream>>>(W, Wt, K, N);
    };
    auto cvt = [&](const float* A, _Float16* O, size_t n) {
        int n4 = (int)(n / 4);
        cvt_f16_kernel<<<(n4 + 255) / 256, 256, 0, stream>>>(A, O, n4);
    };
    auto gemm = [&](const _Float16* A16, const _Float16* Bt, float* C,
                    int M, int K, int N, int relu) {
        dim3 grid(N / 64, M / 128);
        gemm_wmma_f16_kernel<<<grid, 256, 0, stream>>>(A16, Bt, C, M, K, N, relu);
    };
    auto ln = [&](const float* A, const float* R, const float* G, const float* Bb, float* O) {
        add_ln_kernel<<<NM, 256, 0, stream>>>(A, R, G, Bb, O);
    };
    auto attn = [&](const float* Q, const float* Kk, const float* V, float* O, int t) {
        attn_kernel<<<NB * NH, 64, 0, stream>>>(Q, Kk, V, O, ends, t);
    };

    // ---- convert + transpose all weights to f16 [N,K] ----
    conv(eWin, wt_eWin, NE, ND);
    conv(eWq,  wt_eWq,  ND, ND);
    conv(eWk,  wt_eWk,  ND, ND);
    conv(eWv,  wt_eWv,  ND, ND);
    conv(eWo,  wt_eWo,  ND, ND);
    conv(eF1,  wt_eF1,  ND, NF);
    conv(eF2,  wt_eF2,  NF, ND);
    conv(cWq,  wt_cWq,  ND, ND);
    conv(cWk,  wt_cWk,  ND, ND);
    conv(cWv,  wt_cWv,  ND, ND);
    conv(cWo,  wt_cWo,  ND, ND);
    conv(cF1,  wt_cF1,  ND, NF);
    conv(cF2,  wt_cF2,  NF, ND);

    const size_t NMD = (size_t)NM * ND;
    const size_t NMF = (size_t)NM * NF;

    // ---- recurrent loop over segments ----
    for (int t = 0; t < NT; ++t) {
        // ----- encoder on segment t -----
        gather_seg_kernel<<<NM, 256, 0, stream>>>(seqs, Xb, t);
        cvt(Xb, H16A, NMD);
        gemm(H16A, wt_eWin, Hb, NM, NE, ND, 0);        // h = x @ eWin
        cvt(Hb, H16B, NMD);                             // shared by Q/K/V
        gemm(H16B, wt_eWq, Qb, NM, ND, ND, 0);
        gemm(H16B, wt_eWk, Kb, NM, ND, ND, 0);
        gemm(H16B, wt_eWv, Vb, NM, ND, ND, 0);
        attn(Qb, Kb, Vb, ATTNb, t);                    // masked self-attn
        cvt(ATTNb, H16A, NMD);
        gemm(H16A, wt_eWo, T1b, NM, ND, ND, 0);
        ln(Hb, T1b, eG1, eB1, T2b);                    // h1 = LN(h + attn)
        cvt(T2b, H16A, NMD);
        gemm(H16A, wt_eF1, MIDb, NM, ND, NF, 1);       // relu(h1 @ eF1)
        cvt(MIDb, H16A, NMF);
        gemm(H16A, wt_eF2, T1b, NM, NF, ND, 0);
        float* enc_dst = (t == 0) ? out : ENCb;        // t==0: state = encoded
        ln(T2b, T1b, eG2, eB2, enc_dst);               // enc = LN(h1 + ffn)

        if (t > 0) {
            // ----- connection: curr=ENCb cross-attends over prev=out -----
            cvt(ENCb, H16A, NMD);
            gemm(H16A, wt_cWq, Qb, NM, ND, ND, 0);
            cvt(out, H16B, NMD);                        // shared by K/V
            gemm(H16B, wt_cWk, Kb, NM, ND, ND, 0);
            gemm(H16B, wt_cWv, Vb, NM, ND, ND, 0);
            attn(Qb, Kb, Vb, ATTNb, -1);               // prev_lens = SEG (full)
            cvt(ATTNb, H16A, NMD);
            gemm(H16A, wt_cWo, T1b, NM, ND, ND, 0);
            ln(ENCb, T1b, cG1, cB1, T2b);              // h1 = LN(curr + attn)
            cvt(T2b, H16A, NMD);
            gemm(H16A, wt_cF1, MIDb, NM, ND, NF, 1);
            cvt(MIDb, H16A, NMF);
            gemm(H16A, wt_cF2, T1b, NM, NF, ND, 0);
            ln(T2b, T1b, cG2, cB2, out);               // state = LN(h1 + ffn)
        }
    }
}
